// STTransformerLayer_50766513439027
// MI455X (gfx1250) — compile-verified, matrix operand order checked by reading, not run on hardware
//
#include <hip/hip_runtime.h>

typedef __attribute__((ext_vector_type(16))) _Float16 v16h;
typedef __attribute__((ext_vector_type(8)))  _Float16 v8h;
typedef __attribute__((ext_vector_type(8)))  float    v8f;

#define EE 512
#define BB 16
#define JJ 24
#define WW 120

// ---------------- WMMA / async helpers ----------------

__device__ __forceinline__ v8f wmma32(v16h a, v16h b, v8f c) {
  // D = A(16x32 f16) * B(32x16 f16) + C(16x16 f32)
  return __builtin_amdgcn_wmma_f32_16x16x32_f16(false, a, false, b, (short)0, c,
                                                false, false);
}

// CDNA5 async global->LDS copy of 16 bytes per lane (ASYNCcnt-tracked).
__device__ __forceinline__ void async_copy16(unsigned int lds_off,
                                             const _Float16* gp) {
  asm volatile("global_load_async_to_lds_b128 %0, %1, off"
               :: "v"(lds_off), "v"((unsigned long long)(uintptr_t)gp)
               : "memory");
}
__device__ __forceinline__ void wait_async0() {
  asm volatile("s_wait_asynccnt 0" ::: "memory");
}
__device__ __forceinline__ unsigned int lds_off32(const void* p) {
  return (unsigned int)(uintptr_t)p;  // low 32 bits of LDS aperture address
}

// Load one 16x32 f16 operand tile from LDS (row-major, row stride `ld` halves).
// CDNA5 layout: lanes 0-15 hold row m=lane, K in {0..7,16..23};
// lanes 16-31 hold row m=lane-16, K in {8..15,24..31}.
__device__ __forceinline__ v16h ld_frag(const _Float16* p, int ld) {
  const int lane = threadIdx.x & 31;
  const _Float16* q = p + (lane & 15) * ld + ((lane >> 4) << 3);
  v8h lo = *(const v8h*)q;
  v8h hi = *(const v8h*)(q + 16);
  v16h f;
#pragma unroll
  for (int i = 0; i < 8; ++i) { f[i] = lo[i]; f[i + 8] = hi[i]; }
  return f;
}

// Stage/convert 8 contiguous elements (f32 -> f16).
__device__ __forceinline__ void stage8(_Float16* dst, const float* src) {
  const float4 f0 = ((const float4*)src)[0];
  const float4 f1 = ((const float4*)src)[1];
  v8h o;
  o[0] = (_Float16)f0.x; o[1] = (_Float16)f0.y;
  o[2] = (_Float16)f0.z; o[3] = (_Float16)f0.w;
  o[4] = (_Float16)f1.x; o[5] = (_Float16)f1.y;
  o[6] = (_Float16)f1.z; o[7] = (_Float16)f1.w;
  *(v8h*)dst = o;
}

// ---------------- f32 -> f16 repack kernels ----------------

// xw[j][w*16+b][e] = x[lo+w][b][j][e], lo = max(0, t-120)
__global__ __launch_bounds__(256) void gather_xw_kernel(
    const float* __restrict__ x, const int* __restrict__ tptr,
    _Float16* __restrict__ xw) {
  const long tid = (long)blockIdx.x * 256 + threadIdx.x;  // 2,949,120 total
  const int t = *tptr;
  const int lo = (t > WW) ? (t - WW) : 0;
  const long e0 = (tid & 63) << 3;
  const long rem = tid >> 6;            // j*1920 + (w*16+b)
  const long r = rem % 1920;
  const long j = rem / 1920;
  const int w = (int)(r >> 4);
  const int b = (int)(r & 15);
  const float* src = x + ((((long)(lo + w) * 16 + b) * 24 + j) << 9) + e0;
  stage8(xw + (rem << 9) + e0, src);
}

// sp[(j*16+b)][e] = x[t][b][j][e]
__global__ __launch_bounds__(256) void gather_sp_kernel(
    const float* __restrict__ x, const int* __restrict__ tptr,
    _Float16* __restrict__ sp) {
  const long tid = (long)blockIdx.x * 256 + threadIdx.x;  // 24576 total
  const int t = *tptr;
  const long e0 = (tid & 63) << 3;
  const long r = tid >> 6;              // j*16+b
  const long j = r >> 4;
  const long b = r & 15;
  const float* src = x + ((((long)t * 16 + b) * 24 + j) << 9) + e0;
  stage8(sp + (r << 9) + e0, src);
}

// plain f32 -> f16 conversion (weights), 8 elements/thread
__global__ __launch_bounds__(256) void f32to16_kernel(
    const float* __restrict__ src, _Float16* __restrict__ dst) {
  const long i = ((long)blockIdx.x * 256 + threadIdx.x) << 3;
  stage8(dst + i, src + i);
}

// ---------------- NT GEMM: C[m,n] = sum_k A[m,k]*W16[n,k] + bias[n] --------------
// 128x64 block tile, 8 waves each 32x32 (4 WMMA / k-step), double-buffered LDS
// filled with async global->LDS b128 copies.

template <typename OT, bool RELU2>
__global__ __launch_bounds__(256) void gemm_kernel(
    const _Float16* __restrict__ A, int lda, long zsA,
    const _Float16* __restrict__ W16, long zsW,
    const float* __restrict__ bias, long zsB,
    OT* __restrict__ C, int ldc, long zsC, int K) {
  __shared__ _Float16 As[2][128 * 40];  // 128 rows x 32 k (+8 pad)
  __shared__ _Float16 Bs[2][64 * 40];
  const int z = blockIdx.z;
  const long row0 = (long)blockIdx.y * 128;
  const long col0 = (long)blockIdx.x * 64;
  const int tid = threadIdx.x;
  const int r4 = tid >> 2;           // 0..63
  const int ac = (tid & 3) << 3;     // 0,8,16,24
  const _Float16* aP0 = A + zsA * z + (row0 + r4) * (long)lda + ac;
  const _Float16* aP1 = aP0 + 64 * (long)lda;
  const _Float16* wP  = W16 + zsW * z + (col0 + r4) * (long)K + ac;
  unsigned int ldsA0[2], ldsA1[2], ldsB[2];
#pragma unroll
  for (int s = 0; s < 2; ++s) {
    ldsA0[s] = lds_off32(&As[s][r4 * 40 + ac]);
    ldsA1[s] = lds_off32(&As[s][(r4 + 64) * 40 + ac]);
    ldsB[s]  = lds_off32(&Bs[s][r4 * 40 + ac]);
  }
  const int wv = tid >> 5;
  const int wm = wv & 3;             // 4 row-waves (32 rows each)
  const int wn = wv >> 2;            // 2 col-waves (32 cols each)
  v8f acc00 = {}, acc01 = {}, acc10 = {}, acc11 = {};
  const int ksteps = K >> 5;
  // prologue: stage 0
  async_copy16(ldsA0[0], aP0);
  async_copy16(ldsA1[0], aP1);
  async_copy16(ldsB[0], wP);
  aP0 += 32; aP1 += 32; wP += 32;
  for (int kt = 0; kt < ksteps; ++kt) {
    const int cur = kt & 1;
    wait_async0();                   // current stage resident (this wave)
    __syncthreads();                 // all waves resident + done reading next buf
    if (kt + 1 < ksteps) {           // issue next stage; overlaps compute below
      const int nb = cur ^ 1;
      async_copy16(ldsA0[nb], aP0);
      async_copy16(ldsA1[nb], aP1);
      async_copy16(ldsB[nb], wP);
      aP0 += 32; aP1 += 32; wP += 32;
    }
    const _Float16* Ac = As[cur];
    const _Float16* Bc = Bs[cur];
    v16h a0 = ld_frag(&Ac[(wm * 32) * 40], 40);
    v16h a1 = ld_frag(&Ac[(wm * 32 + 16) * 40], 40);
    v16h b0 = ld_frag(&Bc[(wn * 32) * 40], 40);
    v16h b1 = ld_frag(&Bc[(wn * 32 + 16) * 40], 40);
    acc00 = wmma32(a0, b0, acc00);
    acc01 = wmma32(a0, b1, acc01);
    acc10 = wmma32(a1, b0, acc10);
    acc11 = wmma32(a1, b1, acc11);
  }
  const int lane = tid & 31;
  const int rsub = (lane >> 4) << 3;
  const int nsub = lane & 15;
#pragma unroll
  for (int fr = 0; fr < 2; ++fr) {
#pragma unroll
    for (int fc = 0; fc < 2; ++fc) {
      const v8f cc = fr ? (fc ? acc11 : acc10) : (fc ? acc01 : acc00);
      const long gn = col0 + wn * 32 + fc * 16 + nsub;
      const float bv = bias[zsB * z + gn];
#pragma unroll
      for (int i = 0; i < 8; ++i) {
        const long gm = row0 + wm * 32 + fr * 16 + rsub + i;
        float v = cc[i] + bv;
        if (RELU2) v = fmaxf(v, 0.f) * 2.f;
        C[zsC * z + gm * (long)ldc + gn] = (OT)v;
      }
    }
  }
}

// ---------------- attention: O = softmax(QK^T/8 + mask) V, per (z,b,h) ------------

__global__ __launch_bounds__(256) void attn_kernel(
    const _Float16* __restrict__ qkv, _Float16* __restrict__ outp,
    int L, int causal, long zq, long zo) {
  __shared__ _Float16 Qs[128 * 64];   // Q rows x 64, later reused as V^T [64][128]
  __shared__ _Float16 Ks[128 * 64];
  __shared__ _Float16 SP[128 * 128];  // scores then probabilities (f16)
  const int h = blockIdx.x, bb = blockIdx.y, z = blockIdx.z;
  const int tid = threadIdx.x;
  const long qb = (long)z * zq + (long)bb * (3 * EE) + (long)h * 64;
  const int RS = BB * 3 * EE;         // 24576 halves between rows
  const int nt = (L + 15) >> 4;
  // stage Q,K via async copies (zero-pad rows >= L with plain ds stores)
#pragma unroll
  for (int it = 0; it < 4; ++it) {
    const int idx = tid + 256 * it;
    const int r = idx >> 3;
    const int c0v = (idx & 7) << 3;
    if (r < L) {
      const _Float16* p = qkv + qb + (long)r * RS + c0v;
      async_copy16(lds_off32(&Qs[r * 64 + c0v]), p);
      async_copy16(lds_off32(&Ks[r * 64 + c0v]), p + EE);
    } else {
      v8h zz;
#pragma unroll
      for (int i = 0; i < 8; ++i) zz[i] = (_Float16)0.f;
      *(v8h*)&Qs[r * 64 + c0v] = zz;
      *(v8h*)&Ks[r * 64 + c0v] = zz;
    }
  }
  wait_async0();
  __syncthreads();
  const int wv = tid >> 5;
  const int lane = tid & 31;
  // scores S = Q K^T * 0.125. For causal attention skip tiles strictly above the
  // diagonal: their entries only feed masked positions, and the softmax pass
  // rewrites every P entry (zeros where masked) before the O-GEMM reads SP.
  const int ntt = nt * nt;
  for (int tix = wv; tix < ntt; tix += 8) {
    const int rt = tix / nt, ct = tix - rt * nt;
    if (causal && ct > rt) continue;
    v16h a0 = ld_frag(&Qs[rt * 16 * 64], 64);
    v16h a1 = ld_frag(&Qs[rt * 16 * 64 + 32], 64);
    v16h b0 = ld_frag(&Ks[ct * 16 * 64], 64);
    v16h b1 = ld_frag(&Ks[ct * 16 * 64 + 32], 64);
    v8f c = {};
    c = wmma32(a0, b0, c);
    c = wmma32(a1, b1, c);
    const int m0 = rt * 16 + ((lane >> 4) << 3);
    const int cn = ct * 16 + (lane & 15);
#pragma unroll
    for (int i = 0; i < 8; ++i)
      SP[(m0 + i) * 128 + cn] = (_Float16)(c[i] * 0.125f);
  }
  __syncthreads();
  // stage V^T into Qs region (zero-pad keys >= L)
  _Float16* Vt = Qs;
#pragma unroll
  for (int it = 0; it < 4; ++it) {
    const int idx = tid + 256 * it;
    const int kk = idx >> 3;
    const int n0 = (idx & 7) << 3;
    v8h v;
    if (kk < L) {
      v = *(const v8h*)(qkv + qb + 2 * EE + (long)kk * RS + n0);
    } else {
#pragma unroll
      for (int i = 0; i < 8; ++i) v[i] = (_Float16)0.f;
    }
#pragma unroll
    for (int i = 0; i < 8; ++i) Vt[(n0 + i) * 128 + kk] = v[i];
  }
  // masked softmax, one row per thread (writes ALL 128 cols: zeros where masked)
  if (tid < 128) {
    const int r = tid;
    const int valid = (r < L) ? (causal ? (r + 1) : L) : 0;
    float mx = -3.0e38f;
    for (int c2 = 0; c2 < valid; ++c2) mx = fmaxf(mx, (float)SP[r * 128 + c2]);
    float s = 0.f;
    for (int c2 = 0; c2 < valid; ++c2) s += __expf((float)SP[r * 128 + c2] - mx);
    const float inv = (valid > 0) ? (1.f / s) : 0.f;
    for (int c2 = 0; c2 < 128; ++c2) {
      const float pv =
          (c2 < valid) ? __expf((float)SP[r * 128 + c2] - mx) * inv : 0.f;
      SP[r * 128 + c2] = (_Float16)pv;
    }
  }
  __syncthreads();
  // O = P V  (K-dim = nt*16, zero padded)
  const long ob = (long)z * zo + (long)bb * EE + (long)h * 64;
  const int otiles = nt * 4;
  const int ks2 = (nt * 16) >> 5;
  for (int tix = wv; tix < otiles; tix += 8) {
    const int rt = tix >> 2, cn = tix & 3;
    v8f c = {};
    for (int kk = 0; kk < ks2; ++kk) {
      v16h a = ld_frag(&SP[rt * 16 * 128 + kk * 32], 128);
      v16h b = ld_frag(&Vt[cn * 16 * 128 + kk * 32], 128);
      c = wmma32(a, b, c);
    }
    const int m0 = rt * 16 + ((lane >> 4) << 3);
    const int n2 = cn * 16 + (lane & 15);
#pragma unroll
    for (int i = 0; i < 8; ++i) {
      const int m = m0 + i;
      if (m < L) outp[ob + (long)m * (BB * EE) + n2] = (_Float16)c[i];
    }
  }
}

// ---------------- add + LayerNorm (one wave per row of 512) ----------------

__global__ __launch_bounds__(256) void ln_temporal_kernel(
    const _Float16* __restrict__ xw, const _Float16* __restrict__ mha,
    const float* __restrict__ g, const float* __restrict__ bta,
    float* __restrict__ outp, _Float16* __restrict__ out16) {
  const int wv = threadIdx.x >> 5, lane = threadIdx.x & 31;
  const long row = (long)blockIdx.x * 8 + wv;  // w*384 + b*24 + j
  const int j = (int)(row % 24);
  const long rem = row / 24;                   // w*16 + b
  const long src = ((long)j * 1920 + rem) * 512;
  float vals[16];
  float s = 0.f, ss = 0.f;
#pragma unroll
  for (int i = 0; i < 16; ++i) {
    const int e = lane + 32 * i;
    const float v = (float)xw[src + e] + (float)mha[src + e];
    vals[i] = v; s += v; ss += v * v;
  }
#pragma unroll
  for (int off = 16; off >= 1; off >>= 1) {
    s += __shfl_xor(s, off, 32);
    ss += __shfl_xor(ss, off, 32);
  }
  const float mean = s * (1.f / 512.f);
  const float var = ss * (1.f / 512.f) - mean * mean;
  const float inv = rsqrtf(var + 1e-5f);
  const long dst = row * 512;
#pragma unroll
  for (int i = 0; i < 16; ++i) {
    const int e = lane + 32 * i;
    const float y = (vals[i] - mean) * inv * g[e] + bta[e];
    outp[dst + e] = y;
    out16[dst + e] = (_Float16)y;
  }
}

__global__ __launch_bounds__(256) void ln_spatial_kernel(
    const _Float16* __restrict__ sp, const _Float16* __restrict__ mha,
    const float* __restrict__ g, const float* __restrict__ bta,
    const int* __restrict__ tptr, float* __restrict__ outp,
    _Float16* __restrict__ out16) {
  const int wv = threadIdx.x >> 5, lane = threadIdx.x & 31;
  const int r = blockIdx.x * 8 + wv;  // j*16+b, 0..383
  const int j = r >> 4, b = r & 15;
  const int t = *tptr;
  const int wt = (t < WW - 1) ? t : (WW - 1);
  const long src = (long)r * 512;
  float vals[16];
  float s = 0.f, ss = 0.f;
#pragma unroll
  for (int i = 0; i < 16; ++i) {
    const int e = lane + 32 * i;
    const float v = (float)sp[src + e] + (float)mha[src + e];
    vals[i] = v; s += v; ss += v * v;
  }
#pragma unroll
  for (int off = 16; off >= 1; off >>= 1) {
    s += __shfl_xor(s, off, 32);
    ss += __shfl_xor(ss, off, 32);
  }
  const float mean = s * (1.f / 512.f);
  const float var = ss * (1.f / 512.f) - mean * mean;
  const float inv = rsqrtf(var + 1e-5f);
  const long dst = (((long)wt * 16 + b) * 24 + j) * 512;
#pragma unroll
  for (int i = 0; i < 16; ++i) {
    const int e = lane + 32 * i;
    const float nv = outp[dst + e] + (vals[i] - mean) * inv * g[e] + bta[e];
    outp[dst + e] = nv;
    out16[dst + e] = (_Float16)nv;
  }
}

// ---------------- host orchestration ----------------

extern "C" void kernel_launch(void* const* d_in, const int* in_sizes, int n_in,
                              void* d_out, int out_size, void* d_ws, size_t ws_size,
                              hipStream_t stream) {
  (void)in_sizes; (void)n_in; (void)out_size; (void)ws_size;
  const float* x        = (const float*)d_in[0];
  const int*   tptr     = (const int*)d_in[1];
  const float* sp_Wqkv  = (const float*)d_in[2];
  const float* sp_bqkv  = (const float*)d_in[3];
  const float* sp_Wo    = (const float*)d_in[4];
  const float* sp_bo    = (const float*)d_in[5];
  const float* sp_ln_g  = (const float*)d_in[6];
  const float* sp_ln_b  = (const float*)d_in[7];
  const float* tp_Wqkv  = (const float*)d_in[8];
  const float* tp_bqkv  = (const float*)d_in[9];
  const float* tp_Wo    = (const float*)d_in[10];
  const float* tp_bo    = (const float*)d_in[11];
  const float* tp_ln_g  = (const float*)d_in[12];
  const float* tp_ln_b  = (const float*)d_in[13];
  const float* W1       = (const float*)d_in[14];
  const float* b1       = (const float*)d_in[15];
  const float* W2       = (const float*)d_in[16];
  const float* b2       = (const float*)d_in[17];
  float* out = (float*)d_out;

  // workspace layout (bytes); total = 294,912,000
  char* ws = (char*)d_ws;
  _Float16* xw16   = (_Float16*)(ws);                 //  47,185,920 (24*1920*512)
  _Float16* qkv_t  = (_Float16*)(ws + 47185920L);     // 141,557,760 (24*1920*1536)
  _Float16* attn_t = (_Float16*)(ws + 188743680L);    //  47,185,920
  _Float16* tmha   = qkv_t;                           // reuse after attention
  _Float16* tout16 = attn_t;                          // reuse after out-proj
  _Float16* hbuf   = (_Float16*)(ws);                 // FFN hidden (188.7MB) reuse
  char* spb = ws + 235929600L;
  _Float16* sp16   = (_Float16*)(spb);                // 393,216
  _Float16* qkv_s  = (_Float16*)(spb + 393216L);      // 1,179,648
  _Float16* attn_s = (_Float16*)(spb + 1572864L);     // 393,216
  _Float16* smha   = (_Float16*)(spb + 1966080L);     // 393,216
  char* wsw = ws + 238288896L;                        // f16 weights
  _Float16* w16_spqkv = (_Float16*)(wsw);             //  1,572,864
  _Float16* w16_tpqkv = (_Float16*)(wsw + 1572864L);  // 37,748,736
  _Float16* w16_spo   = (_Float16*)(wsw + 39321600L); //    524,288
  _Float16* w16_tpo   = (_Float16*)(wsw + 39845888L); // 12,582,912
  _Float16* w16_1     = (_Float16*)(wsw + 52428800L); //  2,097,152
  _Float16* w16_2     = (_Float16*)(wsw + 54525952L); //  2,097,152

  // 0) one-time f32 -> f16 weight conversion
  f32to16_kernel<<<384, 256, 0, stream>>>(sp_Wqkv, w16_spqkv);
  f32to16_kernel<<<9216, 256, 0, stream>>>(tp_Wqkv, w16_tpqkv);
  f32to16_kernel<<<128, 256, 0, stream>>>(sp_Wo, w16_spo);
  f32to16_kernel<<<3072, 256, 0, stream>>>(tp_Wo, w16_tpo);
  f32to16_kernel<<<512, 256, 0, stream>>>(W1, w16_1);
  f32to16_kernel<<<512, 256, 0, stream>>>(W2, w16_2);

  // 1) repack activations to f16
  gather_sp_kernel<<<96, 256, 0, stream>>>(x, tptr, sp16);
  gather_xw_kernel<<<11520, 256, 0, stream>>>(x, tptr, xw16);

  // 2) QKV projections
  gemm_kernel<_Float16, false><<<dim3(24, 3, 1), 256, 0, stream>>>(
      sp16, 512, 0, w16_spqkv, 0, sp_bqkv, 0, qkv_s, 1536, 0, 512);
  gemm_kernel<_Float16, false><<<dim3(24, 15, 24), 256, 0, stream>>>(
      xw16, 512, 1920L * 512, w16_tpqkv, 1536L * 512, tp_bqkv, 1536,
      qkv_t, 1536, 1920L * 1536, 512);

  // 3) attention (spatial L=24 no mask; temporal L=120 causal)
  attn_kernel<<<dim3(8, 16, 1), 256, 0, stream>>>(qkv_s, attn_s, 24, 0, 0, 0);
  attn_kernel<<<dim3(8, 16, 24), 256, 0, stream>>>(qkv_t, attn_t, 120, 1,
                                                   1920L * 1536, 1920L * 512);

  // 4) output projections
  gemm_kernel<_Float16, false><<<dim3(8, 3, 1), 256, 0, stream>>>(
      attn_s, 512, 0, w16_spo, 0, sp_bo, 0, smha, 512, 0, 512);
  gemm_kernel<_Float16, false><<<dim3(8, 15, 24), 256, 0, stream>>>(
      attn_t, 512, 1920L * 512, w16_tpo, 512L * 512, tp_bo, 512,
      tmha, 512, 1920L * 512, 512);

  // 5) residual + LayerNorm -> tout (f32 in d_out, f16 copy for FFN);
  //    then add spatial branch at w = min(119, t)
  ln_temporal_kernel<<<5760, 256, 0, stream>>>(xw16, tmha, tp_ln_g, tp_ln_b,
                                               out, tout16);
  ln_spatial_kernel<<<48, 256, 0, stream>>>(sp16, smha, sp_ln_g, sp_ln_b, tptr,
                                            out, tout16);

  // 6) FFN: h = relu(tout@W1^T + b1)*2 ; out = h@W2^T + b2
  gemm_kernel<_Float16, true><<<dim3(32, 360, 1), 256, 0, stream>>>(
      tout16, 512, 0, w16_1, 0, b1, 0, hbuf, 2048, 0, 512);
  gemm_kernel<float, false><<<dim3(8, 360, 1), 256, 0, stream>>>(
      hbuf, 2048, 0, w16_2, 0, b2, 0, out, 512, 0, 2048);
}
